// CasualWanSelfAttention_23648089932358
// MI455X (gfx1250) — matrix operand order repaired and checked by reading, not compiled, not run on hardware
//
#include <hip/hip_runtime.h>
#include <hip/hip_bf16.h>
#include <math.h>

typedef __bf16 bf16_t;
typedef __attribute__((ext_vector_type(16))) __bf16 v16bf;
typedef __attribute__((ext_vector_type(8)))  __bf16 v8bf;
typedef __attribute__((ext_vector_type(8)))  float  v8f;

#define S_TOK     2304
#define DIM       1536
#define NH        12
#define HD        128
#define FRAME_TOK 256
#define RMS_EPS   1e-6f

// ---------------------------------------------------------------------------
// CDNA5 async global -> LDS copy (16 bytes / lane), tracked by ASYNCcnt.
// GVS addressing: mem = SGPR base + VGPR byte offset. LDS dest offset is the
// low 32 bits of the generic shared-memory address (wave-relative LDS addr).
// ---------------------------------------------------------------------------
__device__ __forceinline__ void async_copy16(void* lds_ptr, const void* gbase,
                                             uint32_t goff_bytes) {
  uint32_t lds_off = (uint32_t)(size_t)lds_ptr;
  asm volatile("global_load_async_to_lds_b128 %0, %1, %2"
               :
               : "v"(lds_off), "v"(goff_bytes), "s"(gbase)
               : "memory");
}
__device__ __forceinline__ void wait_async0() {
  asm volatile("s_wait_asynccnt 0x0" ::: "memory");
}

// ---------------------------------------------------------------------------
// Fragment loaders: contiguous k order (lane half h holds k = 16h..16h+15), same for A and B
// ---------------------------------------------------------------------------
__device__ __forceinline__ v16bf load16(const bf16_t* p) {
  v8bf lo = *(const v8bf*)p;
  v8bf hi = *(const v8bf*)(p + 8);
  v16bf r;
#pragma unroll
  for (int i = 0; i < 8; ++i) { r[i] = lo[i]; r[i + 8] = hi[i]; }
  return r;
}

// A fragment (16x32 bf16, M x K). Source row-major, row stride ld.
// lane<16 : row=lane,    K = kb+0..15
// lane>=16: row=lane-16, K = kb+16..31
__device__ __forceinline__ v16bf loadA(const bf16_t* base, int ld, int row0,
                                       int kb, int lane) {
  int m    = lane & 15;
  int koff = (lane >> 4) * 16;
  const bf16_t* r = base + (size_t)(row0 + m) * ld + kb + koff;
  v8bf lo = *(const v8bf*)r;
  v8bf hi = *(const v8bf*)(r + 8);
  v16bf out;
#pragma unroll
  for (int i = 0; i < 8; ++i) { out[i] = lo[i]; out[i + 8] = hi[i]; }
  return out;
}

// B fragment (32x16 bf16, K x N), element (k,n) = src[(col0+n)*ld + kb + k]
// lane<16 : col=lane,    K = kb+0..15 ; lane>=16: col=lane-16, K = kb+16..31
__device__ __forceinline__ v16bf loadB(const bf16_t* base, int ld, int col0,
                                       int kb, int lane) {
  int n  = lane & 15;
  int ks = kb + (lane >> 4) * 16;
  return load16(base + (size_t)(col0 + n) * ld + ks);
}

// ---------------------------------------------------------------------------
// Prep kernels
// ---------------------------------------------------------------------------
__global__ void f2bf_kernel(const float* __restrict__ src,
                            bf16_t* __restrict__ dst, int n) {
  int i = blockIdx.x * blockDim.x + threadIdx.x;
  int stride = gridDim.x * blockDim.x;
  for (; i < n; i += stride) dst[i] = (bf16_t)src[i];
}

// cos/sin table: token -> 64 angles (22 temporal, 21 height, 21 width bands)
__global__ void rope_table_kernel(const float* __restrict__ freqs,
                                  float* __restrict__ cosT,
                                  float* __restrict__ sinT) {
  int i = blockIdx.x * blockDim.x + threadIdx.x;  // token*64 + j
  if (i >= S_TOK * 64) return;
  int tok = i >> 6, j = i & 63;
  int t  = tok >> 8;
  int y  = (tok >> 4) & 15;
  int xw = tok & 15;
  int row = (j < 22) ? t : ((j < 43) ? y : xw);
  float ang = freqs[row * 64 + j];
  cosT[i] = cosf(ang);
  sinT[i] = sinf(ang);
}

// ---------------------------------------------------------------------------
// WMMA GEMM: C(MxN) = A(MxK,bf16,row-major) * B(NxK,bf16,row-major)^T + bias
// mode 0: f32 out, 2: bf16 transposed (N-major) out.
// Block: 256 thr = 8 waves; wave = 16 rows x 128 cols; block = 128 x 128.
// B tile (128 cols x 32 k, 8 KB) shared by all 8 waves: staged in LDS via
// async global->LDS DMA, double buffered. All 8 B fragments are preloaded
// before the 8-WMMA chain (one ds wait, then back-to-back WMMAs).
// ---------------------------------------------------------------------------
__global__ __launch_bounds__(256)
void gemm_bf16_kernel(const bf16_t* __restrict__ A, const bf16_t* __restrict__ B,
                      int M, int N, int K,
                      float* __restrict__ outF, bf16_t* __restrict__ outB,
                      const float* __restrict__ bias, int mode) {
  __shared__ __align__(16) bf16_t bstage[2][128 * 32];  // [buf][col][k], 16 KB

  int tid  = threadIdx.x;
  int lane = tid & 31;
  int wave = tid >> 5;
  int row0 = blockIdx.x * 128 + wave * 16;
  int col0 = blockIdx.y * 128;

  // stage one 128x32 B tile: 8 KB = 256 threads x 32 B
  auto stageB = [&](int buf, int kb) {
    int col = tid >> 1;            // 0..127
    int kc  = (tid & 1) * 16;      // element offset 0 or 16
    bf16_t* l = &bstage[buf][col * 32 + kc];
    uint32_t go = (uint32_t)((((size_t)(col0 + col)) * K + kb + kc) * 2);
    async_copy16(l,     B, go);
    async_copy16(l + 8, B, go + 16);
  };

  v8f acc[8];
#pragma unroll
  for (int t = 0; t < 8; ++t)
#pragma unroll
    for (int i = 0; i < 8; ++i) acc[t][i] = 0.f;

  stageB(0, 0);
  int nsteps = K / 32;
  for (int it = 0; it < nsteps; ++it) {
    int kb  = it * 32;
    int cur = it & 1;
    wait_async0();        // own async stores done
    __syncthreads();      // all waves: buf[cur] staged, buf[cur^1] free
    if (kb + 32 < K) stageB(cur ^ 1, kb + 32);

    v16bf a = loadA(A, K, row0, kb, lane);
    v16bf b[8];
#pragma unroll
    for (int t = 0; t < 8; ++t)
      b[t] = loadB(&bstage[cur][0], 32, t * 16, 0, lane);  // ds_load burst
#pragma unroll
    for (int t = 0; t < 8; ++t)
      acc[t] = __builtin_amdgcn_wmma_f32_16x16x32_bf16(
          false, a, false, b[t], (short)0, acc[t], false, false);
  }

  int mh = (lane >> 4) * 8;   // C layout: row = i + 8*(lane>=16), col = lane&15
  int n  = lane & 15;
#pragma unroll
  for (int t = 0; t < 8; ++t) {
#pragma unroll
    for (int i = 0; i < 8; ++i) {
      int r = row0 + mh + i;
      int c = col0 + t * 16 + n;
      float v = acc[t][i] + bias[c];
      if (mode == 0)      outF[(size_t)r * N + c] = v;
      else                outB[(size_t)c * M + r] = (bf16_t)v;  // transposed
    }
  }
}

// ---------------------------------------------------------------------------
// RMS-norm over full 1536 row, then RoPE per head (interleaved pairs),
// then scale (1/sqrt(128) folded into Q), emit bf16. One block per token.
// ---------------------------------------------------------------------------
__global__ __launch_bounds__(256)
void norm_rope_kernel(const float* __restrict__ raw, const float* __restrict__ g,
                      const float* __restrict__ cosT, const float* __restrict__ sinT,
                      bf16_t* __restrict__ outb, float outScale) {
  int tok = blockIdx.x;
  const float* row = raw + (size_t)tok * DIM;
  __shared__ float red[256];
  float ss = 0.f;
  for (int i = threadIdx.x; i < DIM; i += 256) { float v = row[i]; ss += v * v; }
  red[threadIdx.x] = ss;
  __syncthreads();
  for (int s = 128; s > 0; s >>= 1) {
    if (threadIdx.x < s) red[threadIdx.x] += red[threadIdx.x + s];
    __syncthreads();
  }
  float rms = rsqrtf(red[0] / (float)DIM + RMS_EPS);

  for (int p = threadIdx.x; p < DIM / 2; p += 256) {
    int hd  = p >> 6;
    int j   = p & 63;
    int idx = hd * HD + 2 * j;
    float re = row[idx]     * rms * g[idx];
    float im = row[idx + 1] * rms * g[idx + 1];
    float c = cosT[tok * 64 + j], s = sinT[tok * 64 + j];
    outb[(size_t)tok * DIM + idx]     = (bf16_t)((re * c - im * s) * outScale);
    outb[(size_t)tok * DIM + idx + 1] = (bf16_t)((re * s + im * c) * outScale);
  }
}

// ---------------------------------------------------------------------------
// Flash attention, frame-block causal. Grid (head=12, qblock=18), 8 waves.
// A q-block of 128 rows lies inside one 256-token frame => the causal key
// extent kend is BLOCK-UNIFORM, so block barriers are safe and the K/V tiles
// (shared by all 8 waves) are staged in LDS via async DMA, double buffered.
// Causal extent is a multiple of 32, so no per-element masking is needed.
// ---------------------------------------------------------------------------
__global__ __launch_bounds__(256)
void attn_kernel(const bf16_t* __restrict__ qb, const bf16_t* __restrict__ kb_,
                 const bf16_t* __restrict__ vT, bf16_t* __restrict__ ob) {
  __shared__ __align__(16) bf16_t kstage[2][32 * 128];  // [buf][token][dim] 16 KB
  __shared__ __align__(16) bf16_t vstage[2][128 * 32];  // [buf][dim][token] 16 KB
  __shared__ __align__(16) bf16_t pshare[8 * 16 * 32];  // per-wave P tile    8 KB

  int head  = blockIdx.x;
  int tid   = threadIdx.x;
  int lane  = tid & 31;
  int wave  = tid >> 5;
  int qrow0 = blockIdx.y * 128 + wave * 16;
  int kend  = ((int)(blockIdx.y >> 1) + 1) * FRAME_TOK;  // block-uniform
  int hoff  = head * HD;
  bf16_t* myp = pshare + wave * 512;

  // stage 32 keys: K tile 8 KB [tok][dim], V tile 8 KB [dim][tok]; 64 B/thread
  auto stageKV = [&](int buf, int kb0) {
    {
      int tok = tid >> 3;            // 0..31
      int dc  = (tid & 7) * 16;      // dim chunk
      bf16_t* l = &kstage[buf][tok * 128 + dc];
      uint32_t go = (uint32_t)((((size_t)(kb0 + tok)) * DIM + hoff + dc) * 2);
      async_copy16(l,     kb_, go);
      async_copy16(l + 8, kb_, go + 16);
    }
    {
      int dim  = tid >> 1;           // 0..127
      int half = tid & 1;            // token half
      bf16_t* l = &vstage[buf][dim * 32 + half * 16];
      uint32_t go = (uint32_t)((((size_t)(hoff + dim)) * S_TOK + kb0 + half * 16) * 2);
      async_copy16(l,     vT, go);
      async_copy16(l + 8, vT, go + 16);
    }
  };

  v16bf qf[4];
#pragma unroll
  for (int kk = 0; kk < 4; ++kk)
    qf[kk] = loadA(qb, DIM, qrow0, hoff + kk * 32, lane);

  v8f o[8];
  float rm[8], rl[8];
#pragma unroll
  for (int t = 0; t < 8; ++t)
#pragma unroll
    for (int i = 0; i < 8; ++i) o[t][i] = 0.f;
#pragma unroll
  for (int i = 0; i < 8; ++i) { rm[i] = -3.0e38f; rl[i] = 0.f; }

  int mh = (lane >> 4) * 8;
  int nc = lane & 15;

  stageKV(0, 0);
  int nsteps = kend / 32;
  for (int it = 0; it < nsteps; ++it) {
    int kb0 = it * 32;
    int cur = it & 1;
    wait_async0();
    __syncthreads();
    if (kb0 + 32 < kend) stageKV(cur ^ 1, kb0 + 32);

    // scores: S(16x32) = Q(16x128) . K^T; preload all K fragments, then WMMAs
    v16bf kf[8];
#pragma unroll
    for (int kk = 0; kk < 4; ++kk) {
      kf[kk * 2]     = loadB(&kstage[cur][0], 128, 0,  kk * 32, lane);
      kf[kk * 2 + 1] = loadB(&kstage[cur][0], 128, 16, kk * 32, lane);
    }
    v8f c0, c1;
#pragma unroll
    for (int i = 0; i < 8; ++i) { c0[i] = 0.f; c1[i] = 0.f; }
#pragma unroll
    for (int kk = 0; kk < 4; ++kk) {
      c0 = __builtin_amdgcn_wmma_f32_16x16x32_bf16(false, qf[kk], false,
                                                   kf[kk * 2], (short)0, c0,
                                                   false, false);
      c1 = __builtin_amdgcn_wmma_f32_16x16x32_bf16(false, qf[kk], false,
                                                   kf[kk * 2 + 1], (short)0, c1,
                                                   false, false);
    }

    // streaming softmax; C-layout row = i + 8*(lane>=16), cols = lane&15
    float p0[8], p1[8], alpha[8];
#pragma unroll
    for (int i = 0; i < 8; ++i) {
      float v = fmaxf(c0[i], c1[i]);
#pragma unroll
      for (int x = 1; x < 16; x <<= 1) v = fmaxf(v, __shfl_xor(v, x, 32));
      float nm = fmaxf(rm[i], v);
      p0[i] = __expf(c0[i] - nm);
      p1[i] = __expf(c1[i] - nm);
      float sblk = p0[i] + p1[i];
#pragma unroll
      for (int x = 1; x < 16; x <<= 1) sblk += __shfl_xor(sblk, x, 32);
      alpha[i] = __expf(rm[i] - nm);
      rl[i] = rl[i] * alpha[i] + sblk;
      rm[i] = nm;
    }
#pragma unroll
    for (int t = 0; t < 8; ++t)
#pragma unroll
      for (int i = 0; i < 8; ++i) o[t][i] *= alpha[i];

    // P: C layout -> per-wave LDS tile -> A fragment (per-wave in-order LDS)
#pragma unroll
    for (int i = 0; i < 8; ++i) {
      int m = i + mh;
      myp[m * 32 + nc]      = (bf16_t)p0[i];
      myp[m * 32 + 16 + nc] = (bf16_t)p1[i];
    }
    asm volatile("s_wait_dscnt 0x0" ::: "memory");
    v16bf pA = loadA(myp, 32, 0, 0, lane);

    // o += P(16x32) @ V(32x128); preload all V fragments, then WMMAs
    v16bf vb[8];
#pragma unroll
    for (int t = 0; t < 8; ++t)
      vb[t] = loadB(&vstage[cur][0], 32, t * 16, 0, lane);
#pragma unroll
    for (int t = 0; t < 8; ++t)
      o[t] = __builtin_amdgcn_wmma_f32_16x16x32_bf16(false, pA, false, vb[t],
                                                     (short)0, o[t], false, false);
  }

#pragma unroll
  for (int i = 0; i < 8; ++i) {
    float inv = 1.0f / rl[i];
#pragma unroll
    for (int t = 0; t < 8; ++t) o[t][i] *= inv;
  }
#pragma unroll
  for (int t = 0; t < 8; ++t)
#pragma unroll
    for (int i = 0; i < 8; ++i) {
      int r = qrow0 + mh + i;
      int c = hoff + t * 16 + nc;
      ob[(size_t)r * DIM + c] = (bf16_t)o[t][i];
    }
}

// ---------------------------------------------------------------------------

extern "C" void kernel_launch(void* const* d_in, const int* in_sizes, int n_in,
                              void* d_out, int out_size, void* d_ws, size_t ws_size,
                              hipStream_t stream) {
  const float* x     = (const float*)d_in[0];
  const float* Wq    = (const float*)d_in[1];
  const float* bq    = (const float*)d_in[2];
  const float* Wk    = (const float*)d_in[3];
  const float* bk    = (const float*)d_in[4];
  const float* Wv    = (const float*)d_in[5];
  const float* bv    = (const float*)d_in[6];
  const float* Wo    = (const float*)d_in[7];
  const float* bo    = (const float*)d_in[8];
  const float* gq    = (const float*)d_in[9];
  const float* gk    = (const float*)d_in[10];
  const float* freqs = (const float*)d_in[11];
  float* out = (float*)d_out;

  const size_t SD = (size_t)S_TOK * DIM;
  const size_t DD = (size_t)DIM * DIM;

  uint8_t* w = (uint8_t*)d_ws;
  bf16_t* xb   = (bf16_t*)w;                 w += SD * 2;
  bf16_t* wqb  = (bf16_t*)w;                 w += DD * 2;
  bf16_t* wkb  = (bf16_t*)w;                 w += DD * 2;
  bf16_t* wvb  = (bf16_t*)w;                 w += DD * 2;
  bf16_t* wob  = (bf16_t*)w;                 w += DD * 2;
  float*  qraw = (float*)w;                  w += SD * 4;
  float*  kraw = (float*)w;                  w += SD * 4;
  bf16_t* qbb  = (bf16_t*)w;                 w += SD * 2;
  bf16_t* kbb  = (bf16_t*)w;                 w += SD * 2;
  bf16_t* vTb  = (bf16_t*)w;                 w += SD * 2;
  bf16_t* obb  = (bf16_t*)w;                 w += SD * 2;
  float*  cosT = (float*)w;                  w += (size_t)S_TOK * 64 * 4;
  float*  sinT = (float*)w;                  w += (size_t)S_TOK * 64 * 4;

  // 1) precision conversion + rope tables
  f2bf_kernel<<<(int)((SD + 255) / 256), 256, 0, stream>>>(x,  xb,  (int)SD);
  f2bf_kernel<<<(int)((DD + 255) / 256), 256, 0, stream>>>(Wq, wqb, (int)DD);
  f2bf_kernel<<<(int)((DD + 255) / 256), 256, 0, stream>>>(Wk, wkb, (int)DD);
  f2bf_kernel<<<(int)((DD + 255) / 256), 256, 0, stream>>>(Wv, wvb, (int)DD);
  f2bf_kernel<<<(int)((DD + 255) / 256), 256, 0, stream>>>(Wo, wob, (int)DD);
  rope_table_kernel<<<(S_TOK * 64 + 255) / 256, 256, 0, stream>>>(freqs, cosT, sinT);

  // 2) QKV projections (x @ W^T + b)
  dim3 ggrid(S_TOK / 128, DIM / 128);
  gemm_bf16_kernel<<<ggrid, 256, 0, stream>>>(xb, wqb, S_TOK, DIM, DIM,
                                              qraw, nullptr, bq, 0);
  gemm_bf16_kernel<<<ggrid, 256, 0, stream>>>(xb, wkb, S_TOK, DIM, DIM,
                                              kraw, nullptr, bk, 0);
  gemm_bf16_kernel<<<ggrid, 256, 0, stream>>>(xb, wvb, S_TOK, DIM, DIM,
                                              nullptr, vTb, bv, 2);   // transposed bf16

  // 3) RMS-norm + RoPE (+ fold 1/sqrt(128) into Q)
  const float qscale = 0.08838834764831845f;  // 1/sqrt(128)
  norm_rope_kernel<<<S_TOK, 256, 0, stream>>>(qraw, gq, cosT, sinT, qbb, qscale);
  norm_rope_kernel<<<S_TOK, 256, 0, stream>>>(kraw, gk, cosT, sinT, kbb, 1.0f);

  // 4) frame-causal flash attention (async-staged K/V, double buffered)
  attn_kernel<<<dim3(NH, S_TOK / 128), 256, 0, stream>>>(qbb, kbb, vTb, obb);

  // 5) output projection (o @ Wo^T + bo) -> fp32 d_out
  gemm_bf16_kernel<<<ggrid, 256, 0, stream>>>(obb, wob, S_TOK, DIM, DIM,
                                              out, nullptr, bo, 0);
}
